// MambaLMHeadModel_36567351558210
// MI455X (gfx1250) — matrix operand order repaired and checked by reading, not compiled, not run on hardware
//
#include <hip/hip_runtime.h>
#include <hip/hip_bf16.h>
#include <stdint.h>

#define D_MODEL   768
#define N_LAYER   8
#define D_INNER   1536
#define HEADDIM   64
#define NHEADS    24
#define D_STATE   128
#define D_CONV    4
#define CONV_DIM  1792      // D_INNER + 2*D_STATE
#define D_IN_PROJ 3352      // 2*D_INNER + 2*D_STATE + NHEADS
#define NPAD      3456      // D_IN_PROJ padded to multiple of 128 (block N-tile)
#define BATCH     4
#define SEQ       2048
#define BL        (BATCH*SEQ)
#define DIM_INP   6
#define DIM_OUT   5
#define EPS_F     1e-5f

typedef __bf16 bf16;
typedef __attribute__((ext_vector_type(16))) __bf16 v16bf;
typedef __attribute__((ext_vector_type(8)))  float  v8f;

#if defined(__has_builtin)
#if __has_builtin(__builtin_amdgcn_global_load_async_to_lds_b128)
#define HAVE_ASYNC_LDS 1
#endif
#endif

#ifdef HAVE_ASYNC_LDS
// builtin signature (from clang diagnostic): (int4 AS1*, int4 AS3*, imm int, imm int)
typedef int vint4_t __attribute__((vector_size(4 * sizeof(int))));
#define AS1P(p)  ((__attribute__((address_space(1))) vint4_t*)(p))
#define AS3P(p)  ((__attribute__((address_space(3))) vint4_t*)(p))
#endif

// ---------- helpers ----------
__device__ __forceinline__ bf16 to_bf16(float f) {
    union { float f; uint32_t u; } a; a.f = f;
    uint32_t r = (a.u + 0x7FFFu + ((a.u >> 16) & 1u)) >> 16;
    union { unsigned short s; bf16 b; } o; o.s = (unsigned short)r;
    return o.b;
}

__device__ __forceinline__ float sigmoidf_(float x) { return 1.f / (1.f + expf(-x)); }

__device__ __forceinline__ void wait_async_lds() {
#ifdef HAVE_ASYNC_LDS
#if __has_builtin(__builtin_amdgcn_s_wait_asynccnt)
    __builtin_amdgcn_s_wait_asynccnt(0);
#else
    asm volatile("s_wait_asynccnt 0x0" ::: "memory");
#endif
#endif
}

__device__ float blockReduceSum(float v) {
    __shared__ float sm[9];
    #pragma unroll
    for (int o = 16; o > 0; o >>= 1) v += __shfl_xor(v, o, 32);
    __syncthreads();                       // protect sm from previous call
    if ((threadIdx.x & 31) == 0) sm[threadIdx.x >> 5] = v;
    __syncthreads();
    int nw = blockDim.x >> 5;
    float t = (threadIdx.x < (unsigned)nw) ? sm[threadIdx.x] : 0.f;
    if ((threadIdx.x >> 5) == 0) {
        #pragma unroll
        for (int o = 4; o > 0; o >>= 1) t += __shfl_xor(t, o, 32);
        if (threadIdx.x == 0) sm[8] = t;
    }
    __syncthreads();
    return sm[8];
}

// ---------- WMMA B fragment loader (contiguous k = 16*half + s, same order as the A fragment) ----------
// B (W row-major [N, K], B[k][n] = W[n][k]), fragment 32x16 bf16.
// lanes 0-15:  col n = nTile+lane, elems 0..15 -> K[kTile+0..15]
// lanes 16-31: col n = nTile+lane-16, elems    -> K[kTile+16..31]
__device__ __forceinline__ v16bf load_frag_B(const bf16* W, int ldw, int nTile, int kTile, int lane) {
    int half = lane >> 4;
    int l    = lane & 15;
    const bf16* p = W + (size_t)(nTile + l) * ldw + kTile + half * 16;
    union { uint4 q[2]; v16bf v; } u;
    u.q[0] = *(const uint4*)(p);
    u.q[1] = *(const uint4*)(p + 8);
    return u.v;
}

// ---------- GEMM: D[M,N](f32, ld=ldd) = A[M,K](bf16) * W[N,K]^T (bf16) ----------
// Block tile: 64 (M) x 128 (N); 4 waves, each 64x32 = 4x2 WMMA tiles.
// A panel 64x32 staged into LDS with double-buffered async copies, shared by all waves.
__global__ __launch_bounds__(128)
void gemm_bf16_kernel(const bf16* __restrict__ A, const bf16* __restrict__ W,
                      float* __restrict__ D, int M, int N, int K, int ldd) {
    __shared__ bf16 As[2][64 * 32];
    int tid  = threadIdx.x;
    int lane = tid & 31;
    int wave = tid >> 5;
    int nBlocks = N / 128;
    int mBlk = blockIdx.x / nBlocks;
    int nBlk = blockIdx.x % nBlocks;
    int m0 = mBlk * 64;
    int n0 = nBlk * 128 + wave * 32;
    int nk = K / 32;

    // staging: panel = 256 chunks of 16B; chunk c: row = c>>2, k8 = c&3
    int c0 = tid, c1 = tid + 128;
    int row0 = c0 >> 2, k80 = c0 & 3;
    int row1 = c1 >> 2, k81 = c1 & 3;

    auto stage = [&](int ki, int buf) {
        const bf16* s0 = A + (size_t)(m0 + row0) * K + ki * 32 + k80 * 8;
        const bf16* s1 = A + (size_t)(m0 + row1) * K + ki * 32 + k81 * 8;
        bf16* d0 = &As[buf][0] + (size_t)c0 * 8;
        bf16* d1 = &As[buf][0] + (size_t)c1 * 8;
#ifdef HAVE_ASYNC_LDS
        __builtin_amdgcn_global_load_async_to_lds_b128(AS1P(s0), AS3P(d0), 0, 0);
        __builtin_amdgcn_global_load_async_to_lds_b128(AS1P(s1), AS3P(d1), 0, 0);
#else
        *(uint4*)d0 = *(const uint4*)s0;
        *(uint4*)d1 = *(const uint4*)s1;
#endif
    };

    v8f acc[4][2] = {};
    stage(0, 0);
    for (int ki = 0; ki < nk; ki++) {
        wait_async_lds();          // my async copies for panel ki have landed
        __syncthreads();           // panel ki visible block-wide; nobody reads panel ki-1 anymore
        if (ki + 1 < nk) stage(ki + 1, (ki + 1) & 1);
        int buf = ki & 1;
        int half = lane >> 4, l = lane & 15;
        v16bf a[4], b[2];
        #pragma unroll
        for (int i = 0; i < 4; i++) {
            // A frag 16x32: lanes 0-15 row=l K[0..15]; lanes 16-31 K[16..31]
            const bf16* p = &As[buf][(i * 16 + l) * 32 + half * 16];
            union { uint4 q[2]; v16bf v; } u;
            u.q[0] = *(const uint4*)(p);
            u.q[1] = *(const uint4*)(p + 8);
            a[i] = u.v;
        }
        #pragma unroll
        for (int j = 0; j < 2; j++) b[j] = load_frag_B(W, K, n0 + j * 16, ki * 32, lane);
        #pragma unroll
        for (int i = 0; i < 4; i++)
            #pragma unroll
            for (int j = 0; j < 2; j++)
                acc[i][j] = __builtin_amdgcn_wmma_f32_16x16x32_bf16(
                    false, a[i], false, b[j], (short)0, acc[i][j], false, false);
    }
    // store: lanes 0-15 -> rows r (VGPR r), lanes 16-31 -> rows r+8
    int l  = lane & 15;
    int hi = lane >> 4;
    #pragma unroll
    for (int i = 0; i < 4; i++) {
        #pragma unroll
        for (int j = 0; j < 2; j++) {
            int mr = m0 + i * 16 + hi * 8;
            int nc = n0 + j * 16 + l;
            #pragma unroll
            for (int r = 0; r < 8; r++)
                D[(size_t)(mr + r) * ldd + nc] = acc[i][j][r];
        }
    }
}

// ---------- weight conversion ----------
__global__ void convert_Win_kernel(const float* __restrict__ W, bf16* __restrict__ out) {
    size_t idx = (size_t)blockIdx.x * blockDim.x + threadIdx.x;
    size_t total = (size_t)N_LAYER * NPAD * D_MODEL;
    if (idx >= total) return;
    int k = (int)(idx % D_MODEL);
    size_t r = idx / D_MODEL;
    int n = (int)(r % NPAD);
    int l = (int)(r / NPAD);
    float v = (n < D_IN_PROJ) ? W[((size_t)l * D_IN_PROJ + n) * D_MODEL + k] : 0.f;
    out[idx] = to_bf16(v);
}

__global__ void convert_Wout_kernel(const float* __restrict__ W, bf16* __restrict__ out) {
    size_t idx = (size_t)blockIdx.x * blockDim.x + threadIdx.x;
    size_t total = (size_t)N_LAYER * D_MODEL * D_INNER;
    if (idx >= total) return;
    out[idx] = to_bf16(W[idx]);
}

// ---------- embedding ----------
__global__ void embed_kernel(const float* __restrict__ x, const float* __restrict__ W_emb,
                             const float* __restrict__ b_emb, float* __restrict__ h) {
    size_t idx = (size_t)blockIdx.x * blockDim.x + threadIdx.x;
    if (idx >= (size_t)BL * D_MODEL) return;
    int d = (int)(idx % D_MODEL);
    size_t t = idx / D_MODEL;
    const float* xr = x + t * DIM_INP;
    float s = b_emb[d];
    #pragma unroll
    for (int i = 0; i < DIM_INP; i++) s += xr[i] * W_emb[d * DIM_INP + i];
    h[idx] = s;
}

// ---------- residual add + rmsnorm + bf16 cast (D_MODEL = 768, block 256) ----------
__global__ __launch_bounds__(256)
void prenorm_kernel(const float* __restrict__ h, float* __restrict__ res,
                    const float* __restrict__ w, bf16* __restrict__ hn) {
    size_t t = blockIdx.x;
    const float* hr = h + t * D_MODEL;
    float* rr = res + t * D_MODEL;
    float v[3], ss = 0.f;
    #pragma unroll
    for (int i = 0; i < 3; i++) {
        int d = threadIdx.x + i * 256;
        float r = hr[d] + rr[d];
        v[i] = r; rr[d] = r; ss += r * r;
    }
    ss = blockReduceSum(ss);
    float scale = rsqrtf(ss / (float)D_MODEL + EPS_F);
    #pragma unroll
    for (int i = 0; i < 3; i++) {
        int d = threadIdx.x + i * 256;
        hn[t * D_MODEL + d] = to_bf16(v[i] * scale * w[d]);
    }
}

// ---------- causal depthwise conv (width 4) + SiLU + split ----------
__global__ void conv_kernel(const float* __restrict__ zx, const float* __restrict__ cw,
                            const float* __restrict__ cb, float* __restrict__ xh,
                            float* __restrict__ Bm, float* __restrict__ Cm) {
    size_t idx = (size_t)blockIdx.x * blockDim.x + threadIdx.x;
    if (idx >= (size_t)BL * CONV_DIM) return;
    int c = (int)(idx % CONV_DIM);
    size_t t = idx / CONV_DIM;
    int l = (int)(t % SEQ);
    int b = (int)(t / SEQ);
    float acc = cb[c];
    #pragma unroll
    for (int k = 0; k < D_CONV; k++) {
        int ll = l - (D_CONV - 1) + k;
        if (ll >= 0)
            acc += zx[((size_t)b * SEQ + ll) * NPAD + D_INNER + c] * cw[c * D_CONV + k];
    }
    float s = acc * sigmoidf_(acc);
    if (c < D_INNER)                xh[t * D_INNER + c] = s;
    else if (c < D_INNER + D_STATE) Bm[t * D_STATE + (c - D_INNER)] = s;
    else                            Cm[t * D_STATE + (c - D_INNER - D_STATE)] = s;
}

// ---------- dt = softplus(raw + bias) ----------
__global__ void dt_kernel(const float* __restrict__ zx, const float* __restrict__ bias,
                          float* __restrict__ dt) {
    size_t idx = (size_t)blockIdx.x * blockDim.x + threadIdx.x;
    if (idx >= (size_t)BL * NHEADS) return;
    int hd = (int)(idx % NHEADS);
    size_t t = idx / NHEADS;
    float v = zx[t * NPAD + (D_INNER + CONV_DIM) + hd] + bias[hd];
    dt[idx] = (v > 20.f) ? v : log1pf(expf(v));
}

// ---------- SSM scan: 1 block per (batch, head), state 64x128 in registers ----------
__global__ __launch_bounds__(256)
void scan_kernel(const float* __restrict__ xh, const float* __restrict__ Bm,
                 const float* __restrict__ Cm, const float* __restrict__ dt,
                 const float* __restrict__ A_log, const float* __restrict__ Dp,
                 float* __restrict__ ys) {
    int b  = blockIdx.x / NHEADS;
    int hd = blockIdx.x % NHEADS;
    int tid = threadIdx.x;
    int p = tid >> 2;          // 0..63  (HEADDIM row)
    int g = tid & 3;           // n-group, each covers 32 states
    float A  = -expf(A_log[hd]);
    float Dv = Dp[hd];
    float hs[32];
    #pragma unroll
    for (int j = 0; j < 32; j++) hs[j] = 0.f;
    __shared__ float sB[D_STATE], sC[D_STATE], sX[HEADDIM], sdt;
    for (int l = 0; l < SEQ; l++) {
        size_t t = (size_t)b * SEQ + l;
        __syncthreads();
        if (tid < 128)      sB[tid]       = Bm[t * D_STATE + tid];
        else                sC[tid - 128] = Cm[t * D_STATE + (tid - 128)];
        if (tid < 64)       sX[tid]       = xh[t * D_INNER + hd * HEADDIM + tid];
        if (tid == 255)     sdt           = dt[t * NHEADS + hd];
        __syncthreads();
        float dtl = sdt;
        float dec = expf(dtl * A);
        float xp  = sX[p];
        float dtx = dtl * xp;
        float accv = 0.f;
        const float* Bg = sB + g * 32;
        const float* Cg = sC + g * 32;
        #pragma unroll
        for (int j = 0; j < 32; j++) {
            float hh = hs[j] * dec + dtx * Bg[j];
            hs[j] = hh;
            accv += hh * Cg[j];
        }
        accv += __shfl_xor(accv, 1, 32);
        accv += __shfl_xor(accv, 2, 32);
        if (g == 0) ys[t * D_INNER + hd * HEADDIM + p] = accv + Dv * xp;
    }
}

// ---------- gated rmsnorm: yf = rmsnorm(ys * silu(z)) * w  -> bf16 ----------
__global__ __launch_bounds__(256)
void gnorm_kernel(const float* __restrict__ ys, const float* __restrict__ zx,
                  const float* __restrict__ w, bf16* __restrict__ yf) {
    size_t t = blockIdx.x;
    const float* yr = ys + t * D_INNER;
    const float* zr = zx + t * NPAD;          // z = first D_INNER cols of zxbcdt
    float v[6], ss = 0.f;
    #pragma unroll
    for (int i = 0; i < 6; i++) {
        int d = threadIdx.x + i * 256;
        float z = zr[d];
        float gv = yr[d] * (z * sigmoidf_(z));
        v[i] = gv; ss += gv * gv;
    }
    ss = blockReduceSum(ss);
    float scale = rsqrtf(ss / (float)D_INNER + EPS_F);
    #pragma unroll
    for (int i = 0; i < 6; i++) {
        int d = threadIdx.x + i * 256;
        yf[t * D_INNER + d] = to_bf16(v[i] * scale * w[d]);
    }
}

// ---------- final: res+h, rmsnorm, head projection, loss accumulation ----------
__global__ __launch_bounds__(256)
void head_kernel(const float* __restrict__ h, const float* __restrict__ res,
                 const float* __restrict__ nfw, const float* __restrict__ W_head,
                 const float* __restrict__ y, float* __restrict__ out,
                 float* __restrict__ lossAcc) {
    size_t t = blockIdx.x;
    __shared__ float sh[D_MODEL];
    float ss = 0.f;
    #pragma unroll
    for (int i = 0; i < 3; i++) {
        int d = threadIdx.x + i * 256;
        float r = h[t * D_MODEL + d] + res[t * D_MODEL + d];
        sh[d] = r; ss += r * r;
    }
    ss = blockReduceSum(ss);
    float scale = rsqrtf(ss / (float)D_MODEL + EPS_F);
    #pragma unroll
    for (int i = 0; i < 3; i++) {
        int d = threadIdx.x + i * 256;
        sh[d] = sh[d] * scale * nfw[d];
    }
    __syncthreads();
    float err = 0.f;
    for (int o = 0; o < DIM_OUT; o++) {
        float ps = 0.f;
        #pragma unroll
        for (int i = 0; i < 3; i++) {
            int d = threadIdx.x + i * 256;
            ps += sh[d] * W_head[o * D_MODEL + d];
        }
        float tot = blockReduceSum(ps);
        if (threadIdx.x == 0) {
            out[t * DIM_OUT + o] = tot;
            float e = tot - y[t * DIM_OUT + o];
            err += e * e;
        }
    }
    if (threadIdx.x == 0) atomicAdd(lossAcc, err);
}

__global__ void loss_fin_kernel(const float* __restrict__ lossAcc, float* __restrict__ out) {
    out[(size_t)BL * DIM_OUT] = lossAcc[0] / (float)((size_t)BL * DIM_OUT);
}

// ---------- host launch ----------
extern "C" void kernel_launch(void* const* d_in, const int* in_sizes, int n_in,
                              void* d_out, int out_size, void* d_ws, size_t ws_size,
                              hipStream_t stream) {
    const float* x            = (const float*)d_in[0];
    const float* y            = (const float*)d_in[1];
    const float* W_emb        = (const float*)d_in[2];
    const float* b_emb        = (const float*)d_in[3];
    const float* W_in         = (const float*)d_in[4];
    const float* conv_w       = (const float*)d_in[5];
    const float* conv_b       = (const float*)d_in[6];
    const float* dt_bias      = (const float*)d_in[7];
    const float* A_log        = (const float*)d_in[8];
    const float* Dp           = (const float*)d_in[9];
    const float* norm_w       = (const float*)d_in[10];
    const float* W_out        = (const float*)d_in[11];
    const float* block_norm_w = (const float*)d_in[12];
    const float* norm_f_w     = (const float*)d_in[13];
    const float* W_head       = (const float*)d_in[14];

    char* ws = (char*)d_ws;
    size_t off = 0;
    auto take = [&](size_t bytes) { char* p = ws + off; off = (off + bytes + 255) & ~(size_t)255; return p; };

    float* h      = (float*)take((size_t)BL * D_MODEL * 4);
    float* res    = (float*)take((size_t)BL * D_MODEL * 4);
    bf16*  hn     = (bf16*) take((size_t)BL * D_MODEL * 2);
    float* zx     = (float*)take((size_t)BL * NPAD * 4);
    float* xh     = (float*)take((size_t)BL * D_INNER * 4);
    float* Bm     = (float*)take((size_t)BL * D_STATE * 4);
    float* Cm     = (float*)take((size_t)BL * D_STATE * 4);
    float* dt     = (float*)take((size_t)BL * NHEADS * 4);
    float* ys     = (float*)take((size_t)BL * D_INNER * 4);
    bf16*  yf     = (bf16*) take((size_t)BL * D_INNER * 2);
    bf16*  WinB   = (bf16*) take((size_t)N_LAYER * NPAD * D_MODEL * 2);
    bf16*  WoutB  = (bf16*) take((size_t)N_LAYER * D_MODEL * D_INNER * 2);
    float* lossAcc= (float*)take(256);

    (void)hipMemsetAsync(res, 0, (size_t)BL * D_MODEL * 4, stream);
    (void)hipMemsetAsync(lossAcc, 0, 4, stream);

    {
        size_t n = (size_t)N_LAYER * NPAD * D_MODEL;
        convert_Win_kernel<<<(unsigned)((n + 255) / 256), 256, 0, stream>>>(W_in, WinB);
        size_t m = (size_t)N_LAYER * D_MODEL * D_INNER;
        convert_Wout_kernel<<<(unsigned)((m + 255) / 256), 256, 0, stream>>>(W_out, WoutB);
    }
    {
        size_t n = (size_t)BL * D_MODEL;
        embed_kernel<<<(unsigned)((n + 255) / 256), 256, 0, stream>>>(x, W_emb, b_emb, h);
    }

    for (int l = 0; l < N_LAYER; l++) {
        prenorm_kernel<<<BL, 256, 0, stream>>>(h, res, block_norm_w + (size_t)l * D_MODEL, hn);

        // zx[BL, NPAD] = hn[BL, 768] @ WinB[l]^T
        gemm_bf16_kernel<<<(BL / 64) * (NPAD / 128), 128, 0, stream>>>(
            hn, WinB + (size_t)l * NPAD * D_MODEL, zx, BL, NPAD, D_MODEL, NPAD);

        {
            size_t n = (size_t)BL * CONV_DIM;
            conv_kernel<<<(unsigned)((n + 255) / 256), 256, 0, stream>>>(
                zx, conv_w + (size_t)l * CONV_DIM * D_CONV, conv_b + (size_t)l * CONV_DIM,
                xh, Bm, Cm);
        }
        {
            size_t n = (size_t)BL * NHEADS;
            dt_kernel<<<(unsigned)((n + 255) / 256), 256, 0, stream>>>(
                zx, dt_bias + (size_t)l * NHEADS, dt);
        }

        scan_kernel<<<BATCH * NHEADS, 256, 0, stream>>>(
            xh, Bm, Cm, dt, A_log + (size_t)l * NHEADS, Dp + (size_t)l * NHEADS, ys);

        gnorm_kernel<<<BL, 256, 0, stream>>>(ys, zx, norm_w + (size_t)l * D_INNER, yf);

        // h[BL, 768] = yf[BL, 1536] @ WoutB[l]^T
        gemm_bf16_kernel<<<(BL / 64) * (D_MODEL / 128), 128, 0, stream>>>(
            yf, WoutB + (size_t)l * D_MODEL * D_INNER, h, BL, D_MODEL, D_INNER, D_MODEL);
    }

    head_kernel<<<BL, 256, 0, stream>>>(h, res, norm_f_w, W_head, y, (float*)d_out, lossAcc);
    loss_fin_kernel<<<1, 1, 0, stream>>>(lossAcc, (float*)d_out);
}
